// SpeakerModelEC_71579924955432
// MI455X (gfx1250) — matrix operand order repaired and checked by reading, not compiled, not run on hardware
//
#include <hip/hip_runtime.h>
#include <hip/hip_bf16.h>
#include <math.h>

// ---------------- model dimensions ----------------
#define BB   64      // batch
#define EE   512     // embedding dim
#define HH   512     // hidden dim
#define AA   512     // attention dim
#define IMGD 2048    // image feature dim
#define VV   30000   // vocab
#define ML   30      // max len
#define TOK_PAD  0
#define TOK_SOS  1
#define TOK_EOS  2
#define TOK_NOHS 3

typedef __attribute__((ext_vector_type(16))) __bf16 v16bf;
typedef __attribute__((ext_vector_type(8)))  __bf16 v8bf;
typedef __attribute__((ext_vector_type(8)))  float  v8f;

// ============================================================
// WMMA bf16 K-loop over fp32 sources (convert in registers).
// A fragment: lanes 0-15 hold K 0..15,
// lanes 16-31 hold K 16..31.  B fragment: lane%16 = n,
// lanes 0-15 K 0..15, lanes 16-31 K 16..31 (contiguous in W[N,K]).
// ============================================================
__device__ __forceinline__ v8f wmma_k_loop_f32(v8f acc,
                                               const float* __restrict__ arow,
                                               const float* __restrict__ wrow,
                                               int K, int kA)
{
    for (int k0 = 0; k0 < K; k0 += 32) {
        const float* ap = arow + k0 + kA;
        const float* wp = wrow + k0;
        __builtin_prefetch(wp + 64, 0, 1);   // global_prefetch_b8 on W stream
        v16bf af, bf;
#pragma unroll
        for (int i = 0; i < 8; ++i)  af[i]     = (__bf16)ap[i];
#pragma unroll
        for (int i = 0; i < 8; ++i)  af[i + 8] = (__bf16)ap[i + 8];
#pragma unroll
        for (int i = 0; i < 16; ++i) bf[i]     = (__bf16)wp[i];
        acc = __builtin_amdgcn_wmma_f32_16x16x32_bf16(
                  false, af, false, bf, (short)0, acc, false, false);
    }
    return acc;
}

// ============================================================
// C[64,N] = act( A[64,K] @ W[N,K]^T (+ A2[64,K2] @ W2[N,K2]^T)
//               + bias (+ bias2) )   -- fp32 sources
// One wave -> one 16x16 tile. grid = (ceil(N/16/8), 4), block = 256.
// act: 0=none, 1=relu, 2=tanh
// ============================================================
__launch_bounds__(256)
__global__ void wmma_gemm_kernel(const float* __restrict__ A,  int lda,
                                 const float* __restrict__ W,  int ldw,
                                 const float* __restrict__ A2, int lda2,
                                 const float* __restrict__ W2, int ldw2,
                                 const float* __restrict__ bias,
                                 const float* __restrict__ bias2,
                                 float* __restrict__ C, int ldc,
                                 int N, int K, int K2, int act)
{
    const int lane   = threadIdx.x & 31;
    const int wave   = threadIdx.x >> 5;
    const int ntile  = blockIdx.x * 8 + wave;
    const int ntiles = N >> 4;
    if (ntile >= ntiles) return;               // wave-uniform exit: EXEC all-1s inside
    const int mtile  = blockIdx.y;

    const int mrow = (mtile << 4) + (lane & 15);   // A row of this lane
    const int nrow = (ntile << 4) + (lane & 15);   // W row (output column)
    const int kA   = (lane >> 4) * 16;             // A K sub-offset
    const int kB   = (lane >> 4) * 16;             // B K half offset

    v8f acc = {};
    acc = wmma_k_loop_f32(acc, A + (size_t)mrow * lda,
                               W + (size_t)nrow * ldw + kB, K, kA);
    if (A2 != nullptr) {
        acc = wmma_k_loop_f32(acc, A2 + (size_t)mrow * lda2,
                                   W2 + (size_t)nrow * ldw2 + kB, K2, kA);
    }

    float bv = 0.0f;
    const int col = (ntile << 4) + (lane & 15);
    if (bias)  bv += bias[col];
    if (bias2) bv += bias2[col];
    const int rbase = (mtile << 4) + ((lane >> 4) << 3);  // C row = vgpr + 8*(lane/16)
#pragma unroll
    for (int r = 0; r < 8; ++r) {
        float v = acc[r] + bv;
        if (act == 1)      v = fmaxf(v, 0.0f);
        else if (act == 2) v = tanhf(v);
        C[(size_t)(rbase + r) * ldc + col] = v;
    }
}

// ============================================================
// Pure-bf16 GEMM: C[64,N] = A_bf[64,K] @ W_bf[N,K]^T + bias.
// Inner loop is b128 loads + v_wmma only (no conversions).
// Used for the vocab projection with the bf16 weight cache.
// ============================================================
__launch_bounds__(256)
__global__ void wmma_gemm_bf16_kernel(const __bf16* __restrict__ Abf, int lda,
                                      const __bf16* __restrict__ Wbf, int ldw,
                                      const float* __restrict__ bias,
                                      float* __restrict__ C, int ldc,
                                      int N, int K)
{
    const int lane   = threadIdx.x & 31;
    const int wave   = threadIdx.x >> 5;
    const int ntile  = blockIdx.x * 8 + wave;
    const int ntiles = N >> 4;
    if (ntile >= ntiles) return;
    const int mtile  = blockIdx.y;

    const int mrow = (mtile << 4) + (lane & 15);
    const int nrow = (ntile << 4) + (lane & 15);
    const int kA   = (lane >> 4) * 16;
    const int kB   = (lane >> 4) * 16;

    const __bf16* arow = Abf + (size_t)mrow * lda + kA;   // 16B aligned
    const __bf16* wrow = Wbf + (size_t)nrow * ldw + kB;   // 32B aligned

    v8f acc = {};
    for (int k0 = 0; k0 < K; k0 += 32) {
        __builtin_prefetch(wrow + k0 + 64, 0, 1);
        const v8bf alo = *(const v8bf*)(arow + k0);
        const v8bf ahi = *(const v8bf*)(arow + k0 + 8);
        const v16bf bf = *(const v16bf*)(wrow + k0);
        v16bf af;
#pragma unroll
        for (int i = 0; i < 8; ++i) { af[i] = alo[i]; af[i + 8] = ahi[i]; }
        acc = __builtin_amdgcn_wmma_f32_16x16x32_bf16(
                  false, af, false, bf, (short)0, acc, false, false);
    }

    const int col = (ntile << 4) + (lane & 15);
    const float bv = bias ? bias[col] : 0.0f;
    const int rbase = (mtile << 4) + ((lane >> 4) << 3);
#pragma unroll
    for (int r = 0; r < 8; ++r)
        C[(size_t)(rbase + r) * ldc + col] = acc[r] + bv;
}

// ============================================================
// fp32 -> bf16 elementwise (grid-stride)
// ============================================================
__global__ void convert_bf16_kernel(const float* __restrict__ in,
                                    __bf16* __restrict__ out, long n)
{
    long i = (long)blockIdx.x * blockDim.x + threadIdx.x;
    const long stride = (long)gridDim.x * blockDim.x;
    for (; i < n; i += stride) out[i] = (__bf16)in[i];
}

// ============================================================
// Row standardize with ddof=1:  out = (x - mean) / (std + 1e-6)
// ============================================================
__launch_bounds__(256)
__global__ void standardize_kernel(const float* __restrict__ in, int ld_in,
                                   float* __restrict__ out, int ld_out, int n)
{
    __shared__ float s1[256], s2[256];
    const int row = blockIdx.x, tid = threadIdx.x;
    const float* p = in + (size_t)row * ld_in;
    float sum = 0.0f, sq = 0.0f;
    for (int i = tid; i < n; i += 256) { float v = p[i]; sum += v; sq += v * v; }
    s1[tid] = sum; s2[tid] = sq;
    __syncthreads();
    for (int s = 128; s > 0; s >>= 1) {
        if (tid < s) { s1[tid] += s1[tid + s]; s2[tid] += s2[tid + s]; }
        __syncthreads();
    }
    const float mean = s1[0] / (float)n;
    const float var  = (s2[0] - (float)n * mean * mean) / (float)(n - 1);
    const float inv  = 1.0f / (sqrtf(fmaxf(var, 0.0f)) + 1e-6f);
    float* o = out + (size_t)row * ld_out;
    for (int i = tid; i < n; i += 256) o[i] = (p[i] - mean) * inv;
}

// ============================================================
// x[b, e] = emb[ tok[b] (or fixed_tok), e ]
// ============================================================
__global__ void gather_emb_kernel(const float* __restrict__ emb,
                                  const int* __restrict__ tok, int fixed_tok,
                                  float* __restrict__ x)
{
    const int idx = blockIdx.x * blockDim.x + threadIdx.x;
    if (idx >= BB * EE) return;
    const int b = idx >> 9, e = idx & 511;
    const int t = tok ? tok[b] : fixed_tok;
    x[idx] = emb[(size_t)t * EE + e];
}

// ============================================================
// LSTM pointwise: g[b, i|f|g|o] -> h,c  (gate order i,f,g,o)
// ============================================================
__global__ void lstm_pointwise_kernel(const float* __restrict__ g,
                                      const float* __restrict__ c_in, int ld_cin,
                                      float* __restrict__ h_out, int ld_h,
                                      float* __restrict__ c_out, int ld_c)
{
    const int idx = blockIdx.x * blockDim.x + threadIdx.x;
    if (idx >= BB * HH) return;
    const int b = idx >> 9, j = idx & 511;
    const float* gr = g + (size_t)b * 4 * HH;
    const float gi = gr[j], gf = gr[HH + j], gg = gr[2 * HH + j], go = gr[3 * HH + j];
    const float si = 1.0f / (1.0f + expf(-gi));
    const float sf = 1.0f / (1.0f + expf(-gf));
    const float so = 1.0f / (1.0f + expf(-go));
    const float c  = sf * c_in[(size_t)b * ld_cin + j] + si * tanhf(gg);
    const float h  = so * tanhf(c);
    h_out[(size_t)b * ld_h + j] = h;
    c_out[(size_t)b * ld_c + j] = c;
}

// ============================================================
// Seed decoder state: copy dec_hid -> out + h1 + c1; init tok/seen
// ============================================================
__global__ void seed_decoder_kernel(const float* __restrict__ dec_hid,
                                    float* __restrict__ out_dec_hid,
                                    float* __restrict__ h1, int ld_h,
                                    float* __restrict__ c1,
                                    int* __restrict__ tok, int* __restrict__ seen)
{
    const int idx = blockIdx.x * blockDim.x + threadIdx.x;
    if (idx < BB * HH) {
        const int b = idx >> 9, j = idx & 511;
        const float v = dec_hid[idx];
        out_dec_hid[idx] = v;
        h1[(size_t)b * ld_h + j] = v;
        c1[idx] = v;
    }
    if (idx < BB) { tok[idx] = TOK_SOS; seen[idx] = 0; }
}

// ============================================================
// Deterministic token selection (mode of nucleus distribution):
// argmax over V logits; records pre-step EOS mask, writes sentence
// token, advances tok/seen.  One block per batch row.
// ============================================================
__launch_bounds__(256)
__global__ void sample_kernel(const float* __restrict__ logits,
                              int* __restrict__ tok, int* __restrict__ seen,
                              int* __restrict__ maskbuf,
                              float* __restrict__ sentences, int t)
{
    __shared__ float smax[256];
    __shared__ int   sidx[256];
    const int b = blockIdx.x, tid = threadIdx.x;
    const float* p = logits + (size_t)b * VV;
    float best = -3.4e38f; int bi = 0;
    for (int i = tid; i < VV; i += 256) {
        const float v = p[i];
        if (v > best || (v == best && i < bi)) { best = v; bi = i; }
    }
    smax[tid] = best; sidx[tid] = bi;
    __syncthreads();
    for (int s = 128; s > 0; s >>= 1) {
        if (tid < s) {
            const float vo = smax[tid + s]; const int io = sidx[tid + s];
            if (vo > smax[tid] || (vo == smax[tid] && io < sidx[tid])) {
                smax[tid] = vo; sidx[tid] = io;
            }
        }
        __syncthreads();
    }
    if (tid == 0) {
        const int nxt  = sidx[0];
        const int word = nxt % VV;
        const int sp   = seen[b];
        maskbuf[t * BB + b] = sp;                       // mask BEFORE this step
        sentences[b * ML + t] = sp ? (float)TOK_PAD : (float)word;
        seen[b] = sp | (word == TOK_EOS);
        tok[b]  = nxt;
    }
}

// ============================================================
// dec_logits: zero post-EOS steps, then L2-normalize over step axis.
// ============================================================
__global__ void normalize_logits_kernel(float* __restrict__ dec_logits,
                                        const int* __restrict__ maskbuf)
{
    const int idx = blockIdx.x * blockDim.x + threadIdx.x;   // over B*V
    if (idx >= BB * VV) return;
    const int b = idx / VV;
    float vals[ML];
    float ss = 0.0f;
#pragma unroll
    for (int t = 0; t < ML; ++t) {
        float v = maskbuf[t * BB + b] ? 0.0f
                                      : dec_logits[(size_t)t * BB * VV + idx];
        vals[t] = v; ss += v * v;
    }
    const float inv = 1.0f / fmaxf(sqrtf(ss), 1e-12f);
#pragma unroll
    for (int t = 0; t < ML; ++t)
        dec_logits[(size_t)t * BB * VV + idx] = vals[t] * inv;
}

// ============================================================
// host-side orchestration
// ============================================================
extern "C" void kernel_launch(void* const* d_in, const int* in_sizes, int n_in,
                              void* d_out, int out_size, void* d_ws, size_t ws_size,
                              hipStream_t stream)
{
    (void)in_sizes; (void)n_in; (void)out_size;
    // ---- inputs (setup_inputs order, all fp32) ----
    const float* visual_context = (const float*)d_in[0];
    const float* target_img     = (const float*)d_in[1];
    const float* emb            = (const float*)d_in[2];
    const float* W_ih_f = (const float*)d_in[3];  const float* W_hh_f = (const float*)d_in[4];
    const float* b_ih_f = (const float*)d_in[5];  const float* b_hh_f = (const float*)d_in[6];
    const float* W_ih_b = (const float*)d_in[7];  const float* W_hh_b = (const float*)d_in[8];
    const float* b_ih_b = (const float*)d_in[9];  const float* b_hh_b = (const float*)d_in[10];
    const float* W_ih_d = (const float*)d_in[11]; const float* W_hh_d = (const float*)d_in[12];
    const float* b_ih_d = (const float*)d_in[13]; const float* b_hh_d = (const float*)d_in[14];
    const float* W_hid  = (const float*)d_in[15]; const float* b_hid  = (const float*)d_in[16];
    const float* W_dec  = (const float*)d_in[17]; const float* b_dec  = (const float*)d_in[18];
    const float* W_vis  = (const float*)d_in[19]; const float* b_vis  = (const float*)d_in[20];
    const float* W_sep  = (const float*)d_in[21]; const float* b_sep  = (const float*)d_in[22];
    const float* W_hist = (const float*)d_in[23]; const float* b_hist = (const float*)d_in[24];
    const float* W_voc  = (const float*)d_in[25]; const float* b_voc  = (const float*)d_in[26];
    const float* W_e2v  = (const float*)d_in[27]; const float* b_e2v  = (const float*)d_in[28];

    // ---- outputs (flattened tuple: sentences, dec_logits, dec_hid, enc_logits) ----
    float* out            = (float*)d_out;
    float* out_sentences  = out;
    float* out_dec_logits = out + (size_t)BB * ML;
    float* out_dec_hid    = out_dec_logits + (size_t)ML * BB * VV;
    float* out_enc_logits = out_dec_hid + (size_t)BB * HH;

    // ---- workspace layout (fp32 region) ----
    float* wsf = (float*)d_ws;
    float* vc_std   = wsf;
    float* tif_std  = vc_std   + (size_t)BB * 6 * IMGD;
    float* vchtih   = tif_std  + (size_t)BB * IMGD;
    float* cvi      = vchtih   + (size_t)BB * 1024;
    float* hfhb     = cvi      + (size_t)BB * HH;
    float* gbuf     = hfhb     + (size_t)BB * 1024;
    float* dec_hid  = gbuf     + (size_t)BB * 4 * HH;
    float* hist_raw = dec_hid  + (size_t)BB * HH;
    float* h1hist   = hist_raw + (size_t)BB * HH;
    float* c1       = h1hist   + (size_t)BB * 1024;
    float* xbuf     = c1       + (size_t)BB * HH;
    float* ctmp     = xbuf     + (size_t)BB * EE;
    int*   tok      = (int*)(ctmp + (size_t)BB * HH);
    int*   seen     = tok + BB;
    int*   maskbuf  = seen + BB;                      // ML*BB ints

    // ---- optional bf16 weight/activation cache (gated on ws_size) ----
    const size_t base_bytes = ((char*)(maskbuf + ML * BB)) - ((char*)d_ws);
    const size_t bf_off     = (base_bytes + 63) & ~(size_t)63;
    const size_t wvoc_elems = (size_t)VV * 1024;
    const size_t abf_elems  = (size_t)BB * 1024;
    const bool   fast_voc   = ws_size >= bf_off + (wvoc_elems + abf_elems) * 2;
    __bf16* wvoc_bf = (__bf16*)((char*)d_ws + bf_off);
    __bf16* abf     = wvoc_bf + wvoc_elems;

    auto gemm = [&](const float* A, int lda, const float* W, int ldw,
                    const float* A2, int lda2, const float* W2, int ldw2,
                    const float* bias, const float* bias2,
                    float* C, int ldc, int N, int K, int K2, int act) {
        dim3 grid((N / 16 + 7) / 8, 4);
        wmma_gemm_kernel<<<grid, 256, 0, stream>>>(A, lda, W, ldw, A2, lda2, W2, ldw2,
                                                   bias, bias2, C, ldc, N, K, K2, act);
    };

    // ===== partial_forward =====
    standardize_kernel<<<BB, 256, 0, stream>>>(visual_context, 6 * IMGD, vc_std, 6 * IMGD, 6 * IMGD);
    standardize_kernel<<<BB, 256, 0, stream>>>(target_img, IMGD, tif_std, IMGD, IMGD);
    gemm(vc_std, 6 * IMGD, W_vis, 6 * IMGD, nullptr, 0, nullptr, 0,
         b_vis, nullptr, vchtih, 1024, HH, 6 * IMGD, 0, 1);
    gemm(tif_std, IMGD, W_sep, IMGD, nullptr, 0, nullptr, 0,
         b_sep, nullptr, vchtih + HH, 1024, HH, IMGD, 0, 1);
    gemm(vchtih, 1024, W_hid, 1024, nullptr, 0, nullptr, 0,
         b_hid, nullptr, cvi, HH, HH, 1024, 0, 1);
    gather_emb_kernel<<<(BB * EE + 255) / 256, 256, 0, stream>>>(emb, nullptr, TOK_NOHS, xbuf);
    gemm(xbuf, EE, W_ih_f, EE, cvi, HH, W_hh_f, HH,
         b_ih_f, b_hh_f, gbuf, 4 * HH, 4 * HH, EE, HH, 0);
    lstm_pointwise_kernel<<<(BB * HH + 255) / 256, 256, 0, stream>>>(
        gbuf, cvi, HH, hfhb, 1024, ctmp, HH);
    gemm(xbuf, EE, W_ih_b, EE, cvi, HH, W_hh_b, HH,
         b_ih_b, b_hh_b, gbuf, 4 * HH, 4 * HH, EE, HH, 0);
    lstm_pointwise_kernel<<<(BB * HH + 255) / 256, 256, 0, stream>>>(
        gbuf, cvi, HH, hfhb + HH, 1024, ctmp, HH);
    gemm(hfhb, 1024, W_e2v, 1024, nullptr, 0, nullptr, 0,
         b_e2v, nullptr, out_enc_logits, VV, VV, 1024, 0, 0);
    gemm(hfhb, 1024, W_dec, 1024, nullptr, 0, nullptr, 0,
         b_dec, nullptr, dec_hid, HH, HH, 1024, 0, 2);
    gemm(hfhb, 1024, W_hist, 1024, nullptr, 0, nullptr, 0,
         b_hist, nullptr, hist_raw, AA, AA, 1024, 0, 0);
    standardize_kernel<<<BB, 256, 0, stream>>>(hist_raw, AA, h1hist + HH, 1024, AA);
    seed_decoder_kernel<<<(BB * HH + 255) / 256, 256, 0, stream>>>(
        dec_hid, out_dec_hid, h1hist, 1024, c1, tok, seen);

    // one-time bf16 down-convert of W_voc (kept hot in 192MB L2 across 30 steps)
    if (fast_voc) {
        convert_bf16_kernel<<<2048, 256, 0, stream>>>(W_voc, wvoc_bf, (long)wvoc_elems);
    }

    // ===== decode loop (30 steps) =====
    const dim3 voc_grid((VV / 16 + 7) / 8, 4);
    for (int t = 0; t < ML; ++t) {
        float* logits_t = out_dec_logits + (size_t)t * BB * VV;
        gather_emb_kernel<<<(BB * EE + 255) / 256, 256, 0, stream>>>(emb, tok, 0, xbuf);
        gemm(xbuf, EE, W_ih_d, EE, h1hist, 1024, W_hh_d, HH,
             b_ih_d, b_hh_d, gbuf, 4 * HH, 4 * HH, EE, HH, 0);
        lstm_pointwise_kernel<<<(BB * HH + 255) / 256, 256, 0, stream>>>(
            gbuf, c1, HH, h1hist, 1024, c1, HH);
        if (fast_voc) {
            // bf16 activations once per step, then conversion-free WMMA inner loop
            convert_bf16_kernel<<<64, 256, 0, stream>>>(h1hist, abf, (long)abf_elems);
            wmma_gemm_bf16_kernel<<<voc_grid, 256, 0, stream>>>(
                abf, 1024, wvoc_bf, 1024, b_voc, logits_t, VV, VV, 1024);
        } else {
            gemm(h1hist, 1024, W_voc, 1024, nullptr, 0, nullptr, 0,
                 b_voc, nullptr, logits_t, VV, VV, 1024, 0, 0);
        }
        sample_kernel<<<BB, 256, 0, stream>>>(logits_t, tok, seen, maskbuf,
                                              out_sentences, t);
    }

    // ===== post-EOS masking + L2 normalize over step axis =====
    normalize_logits_kernel<<<(BB * VV + 255) / 256, 256, 0, stream>>>(
        out_dec_logits, maskbuf);
}
